// S4Block_73847667687712
// MI455X (gfx1250) — matrix operand order repaired and checked by reading, not compiled, not run on hardware
//
#include <hip/hip_runtime.h>
#include <hip/hip_bf16.h>
#include <stdint.h>

// Problem constants (match reference)
#define BSZ 8
#define TT  4096
#define DM  1024
#define NS  64
#define LN_EPS 1e-5f

typedef __attribute__((ext_vector_type(16))) __bf16 v16bf;
typedef __attribute__((ext_vector_type(8)))  float  v8f;
typedef __attribute__((ext_vector_type(2)))  float  v2f;

// ---------------------------------------------------------------------------
// helpers
// ---------------------------------------------------------------------------
static __device__ __forceinline__ unsigned short f2bf(float f) {
    // round-to-nearest-even float -> bf16 (bit trick)
    unsigned int u = __builtin_bit_cast(unsigned int, f);
    unsigned int r = u + 0x7FFFu + ((u >> 16) & 1u);
    return (unsigned short)(r >> 16);
}

// ---------------------------------------------------------------------------
// Kernel 0: W_out (DM x DM f32) -> bf16
// ---------------------------------------------------------------------------
__global__ __launch_bounds__(256)
void wconv_kernel(const float* __restrict__ W, unsigned short* __restrict__ Wbf) {
    int i4 = (blockIdx.x * 256 + threadIdx.x) * 4;          // DM*DM / 4 threads total
    float4 w = *(const float4*)(W + i4);
    ushort4 o;
    o.x = f2bf(w.x); o.y = f2bf(w.y); o.z = f2bf(w.z); o.w = f2bf(w.w);
    *(ushort4*)(Wbf + i4) = o;
}

// ---------------------------------------------------------------------------
// Kernel 1: LayerNorm over last dim (one (b,t) row per 256-thread block)
// ---------------------------------------------------------------------------
__global__ __launch_bounds__(256)
void ln_kernel(const float* __restrict__ x,
               const float* __restrict__ gamma,
               const float* __restrict__ beta,
               float* __restrict__ xn) {
    const int tid  = threadIdx.x;
    const int lane = tid & 31;
    const int wv   = tid >> 5;
    const size_t base = (size_t)blockIdx.x * DM;

    float4 v = *(const float4*)(x + base + tid * 4);
    float s  = v.x + v.y + v.z + v.w;
    float s2 = v.x * v.x + v.y * v.y + v.z * v.z + v.w * v.w;

    #pragma unroll
    for (int o = 16; o >= 1; o >>= 1) {
        s  += __shfl_xor(s,  o, 32);
        s2 += __shfl_xor(s2, o, 32);
    }

    __shared__ float sm[2][8];
    if (lane == 0) { sm[0][wv] = s; sm[1][wv] = s2; }
    __syncthreads();
    if (tid == 0) {
        float a = 0.f, b = 0.f;
        #pragma unroll
        for (int i = 0; i < 8; ++i) { a += sm[0][i]; b += sm[1][i]; }
        sm[0][0] = a; sm[1][0] = b;
    }
    __syncthreads();

    const float inv = 1.0f / (float)DM;
    float mu   = sm[0][0] * inv;
    float var  = sm[1][0] * inv - mu * mu;
    float rstd = rsqrtf(var + LN_EPS);

    float4 g = *(const float4*)(gamma + tid * 4);
    float4 bb = *(const float4*)(beta + tid * 4);
    float4 o;
    o.x = (v.x - mu) * rstd * g.x + bb.x;
    o.y = (v.y - mu) * rstd * g.y + bb.y;
    o.z = (v.z - mu) * rstd * g.z + bb.z;
    o.w = (v.w - mu) * rstd * g.w + bb.w;
    *(float4*)(xn + base + tid * 4) = o;
}

// ---------------------------------------------------------------------------
// Kernel 2: SSM scan.  4 lanes per (b,d) channel, 16 states per lane.
//           State + discretized params live in VGPRs for all T=4096 steps,
//           stored as float2 so updates lower to V_PK_FMA_F32 (2x f32/instr).
//           Output y written directly as bf16 for the WMMA projection.
// ---------------------------------------------------------------------------
#define NPL 16   // states per lane
#define NP2 8    // packed float2 per lane
#define LPC 4    // lanes per channel

__global__ __launch_bounds__(128)
void scan_kernel(const float* __restrict__ xn,
                 const float* __restrict__ log_A,
                 const float* __restrict__ B_param,
                 const float* __restrict__ C_param,
                 const float* __restrict__ D_param,
                 const float* __restrict__ log_dt,
                 unsigned short* __restrict__ ybf) {
    const int g    = blockIdx.x * 128 + threadIdx.x;   // 0 .. BSZ*DM*LPC-1
    const int b    = g >> 12;                          // / (DM*LPC)
    const int r    = g & 4095;
    const int d    = r >> 2;
    const int part = r & 3;
    const int n0   = part * NPL;

    // Discretize parameters (per lane: 16 states of channel d, packed 2-wide)
    const float dtv = __expf(log_dt[d]);
    v2f dA[NP2], dB[NP2], Cc[NP2], h[NP2];
    #pragma unroll
    for (int i = 0; i < NP2; ++i) {
        int idx = d * NS + n0 + 2 * i;
        dA[i][0] = __expf(-__expf(log_A[idx])     * dtv);
        dA[i][1] = __expf(-__expf(log_A[idx + 1]) * dtv);
        dB[i][0] = B_param[idx]     * dtv;
        dB[i][1] = B_param[idx + 1] * dtv;
        Cc[i][0] = C_param[idx];
        Cc[i][1] = C_param[idx + 1];
        h[i]     = (v2f){0.f, 0.f};
    }
    const float Dp = D_param[d];

    const float* px = xn + ((size_t)b * TT) * DM + d;
    unsigned short* py = ybf + ((size_t)b * TT) * DM + d;

    for (int t = 0; t < TT; ++t) {
        // speculative prefetch of a future timestep's input row
        __builtin_prefetch(px + (size_t)(t + 16) * DM, 0, 0);

        float u = px[(size_t)t * DM];
        v2f uv = (v2f){u, u};
        v2f acc = (v2f){0.f, 0.f};
        #pragma unroll
        for (int i = 0; i < NP2; ++i) {
            // h = dA*h + dB*u   and   acc += C*h   -> V_PK_FMA_F32 pairs
            h[i] = __builtin_elementwise_fma(dA[i], h[i], dB[i] * uv);
            acc  = __builtin_elementwise_fma(Cc[i], h[i], acc);
        }
        float partial = acc[0] + acc[1];
        // reduce across the 4 lanes of this channel
        partial += __shfl_xor(partial, 1, 32);
        partial += __shfl_xor(partial, 2, 32);
        if (part == 0) {
            py[(size_t)t * DM] = f2bf(fmaf(Dp, u, partial));
        }
    }
}

// ---------------------------------------------------------------------------
// Kernel 3: out = residual + Y(bf16) @ W^T(bf16) + b_out   via WMMA bf16
//   Block tile 128(M) x 128(N), K-step 32, double-buffered LDS.
//   8 waves: 4(M) x 2(N); each wave: 32x64 = 2x4 wmma 16x16 tiles.
// ---------------------------------------------------------------------------
union AFrag { v16bf v; uint4 q[2]; };
union CAcc  { v8f  v; float f[8]; };

__global__ __launch_bounds__(256)
void gemm_kernel(const unsigned short* __restrict__ Ybf,   // M x 1024 bf16
                 const unsigned short* __restrict__ Wbf,   // 1024 x 1024 bf16 (row e, col d)
                 const float* __restrict__ resid,          // M x 1024 f32
                 const float* __restrict__ b_out,          // 1024
                 float* __restrict__ out) {                // M x 1024 f32
    __shared__ __align__(16) unsigned short lsA[2][128][32];
    __shared__ __align__(16) unsigned short lsB[2][128][32];

    const int tid  = threadIdx.x;
    const int lane = tid & 31;
    const int wv   = tid >> 5;
    const int wm   = wv & 3;        // 0..3  -> M offset wm*32
    const int wn   = wv >> 2;       // 0..1  -> N offset wn*64
    const int lm   = lane & 15;
    const int lh   = lane >> 4;

    const int m0 = blockIdx.x * 128;
    const int n0 = blockIdx.y * 128;

    auto stage = [&](int bf, int kt) {
        const int k0 = kt * 32;
        #pragma unroll
        for (int i = 0; i < 2; ++i) {
            int c = tid * 2 + i;              // 0..511
            int row = c >> 2;
            int col = (c & 3) * 8;
            *(uint4*)(&lsA[bf][row][col]) =
                *(const uint4*)(Ybf + ((size_t)(m0 + row) << 10) + k0 + col);
        }
        #pragma unroll
        for (int i = 0; i < 2; ++i) {
            int c = tid * 2 + i;
            int row = c >> 2;
            int col = (c & 3) * 8;
            *(uint4*)(&lsB[bf][row][col]) =
                *(const uint4*)(Wbf + ((size_t)(n0 + row) << 10) + k0 + col);
        }
    };

    v8f acc[2][4];
    #pragma unroll
    for (int i = 0; i < 2; ++i)
        #pragma unroll
        for (int j = 0; j < 4; ++j)
            acc[i][j] = (v8f){0.f, 0.f, 0.f, 0.f, 0.f, 0.f, 0.f, 0.f};

    stage(0, 0);
    __syncthreads();

    const int KT = DM / 32;   // 32 K-steps
    for (int kt = 0; kt < KT; ++kt) {
        const int cur = kt & 1;
        if (kt + 1 < KT) stage(cur ^ 1, kt + 1);

        // A fragments: 16x32 bf16, lane<16 holds K{0..7,16..23}, lane>=16 K{8..15,24..31}
        AFrag af[2];
        #pragma unroll
        for (int rr = 0; rr < 2; ++rr) {
            const unsigned short* p = &lsA[cur][wm * 32 + rr * 16 + lm][lh * 8];
            af[rr].q[0] = *(const uint4*)p;          // first 8 K
            af[rr].q[1] = *(const uint4*)(p + 16);   // +16 K offset
        }
        // B fragments: 32x16 bf16, lane<16 col=lm K{0..7,16..23}, lane>=16 col=lm K{8..15,24..31}
        AFrag bf[4];
        #pragma unroll
        for (int ss = 0; ss < 4; ++ss) {
            const unsigned short* p = &lsB[cur][wn * 64 + ss * 16 + lm][lh * 8];
            bf[ss].q[0] = *(const uint4*)p;
            bf[ss].q[1] = *(const uint4*)(p + 16);
        }

        #pragma unroll
        for (int rr = 0; rr < 2; ++rr)
            #pragma unroll
            for (int ss = 0; ss < 4; ++ss)
                acc[rr][ss] = __builtin_amdgcn_wmma_f32_16x16x32_bf16(
                    false, af[rr].v, false, bf[ss].v,
                    (short)0, acc[rr][ss], false, false);

        __syncthreads();
    }

    // Epilogue: fuse bias + residual.  C layout: VGPR e, lanes<16 -> M=e, lanes>=16 -> M=e+8
    #pragma unroll
    for (int rr = 0; rr < 2; ++rr) {
        #pragma unroll
        for (int ss = 0; ss < 4; ++ss) {
            CAcc ca; ca.v = acc[rr][ss];
            const int col = n0 + wn * 64 + ss * 16 + lm;
            const float bo = b_out[col];
            #pragma unroll
            for (int e = 0; e < 8; ++e) {
                const int row = m0 + wm * 32 + rr * 16 + lh * 8 + e;
                const size_t idx = ((size_t)row << 10) + col;
                out[idx] = resid[idx] + ca.f[e] + bo;
            }
        }
    }
}

// ---------------------------------------------------------------------------
// launch
// ---------------------------------------------------------------------------
extern "C" void kernel_launch(void* const* d_in, const int* in_sizes, int n_in,
                              void* d_out, int out_size, void* d_ws, size_t ws_size,
                              hipStream_t stream) {
    const float* x       = (const float*)d_in[0];
    const float* log_A   = (const float*)d_in[1];
    const float* B_param = (const float*)d_in[2];
    const float* C_param = (const float*)d_in[3];
    const float* D_param = (const float*)d_in[4];
    const float* log_dt  = (const float*)d_in[5];
    const float* gamma   = (const float*)d_in[6];
    const float* beta    = (const float*)d_in[7];
    const float* W_out   = (const float*)d_in[8];
    const float* b_out   = (const float*)d_in[9];
    float* out = (float*)d_out;

    const size_t M = (size_t)BSZ * TT;                 // 32768 rows
    const size_t XN_BYTES = M * DM * sizeof(float);    // 134 MB  f32 xn
    const size_t Y_BYTES  = M * DM * sizeof(unsigned short); // 67 MB  bf16 y

    float* xn            = (float*)d_ws;
    unsigned short* ybf  = (unsigned short*)((char*)d_ws + XN_BYTES);
    unsigned short* wbf  = (unsigned short*)((char*)d_ws + XN_BYTES + Y_BYTES);

    // 0) W -> bf16 : DM*DM/1024 blocks
    wconv_kernel<<<(DM * DM) / (256 * 4), 256, 0, stream>>>(W_out, wbf);

    // 1) LayerNorm: one row per block
    ln_kernel<<<(unsigned)M, 256, 0, stream>>>(x, gamma, beta, xn);

    // 2) Scan: BSZ*DM*4 lanes, 128 threads/block
    scan_kernel<<<(BSZ * DM * LPC) / 128, 128, 0, stream>>>(
        xn, log_A, B_param, C_param, D_param, log_dt, ybf);

    // 3) GEMM + residual + bias
    dim3 g((unsigned)(M / 128), DM / 128);
    gemm_kernel<<<g, 256, 0, stream>>>(ybf, wbf, x, b_out, out);
}
